// GAN_4380866642562
// MI455X (gfx1250) — compile-verified, matrix operand order checked by reading, not run on hardware
//
#include <hip/hip_runtime.h>

// ---------------- problem constants ----------------
#define N_NODESC 20000
#define HIDDENC  64
#define HEADSC   4
#define HCC      256          // HEADS*HIDDEN
#define N_EDGESC 640000
#define E_TOTC   660000       // edges + self loops
#define BATCHC   4096
#define NCLSC    10
#define NEG_SLOPEC 0.2f

// ---------------- vector types ----------------
typedef __attribute__((ext_vector_type(16))) __bf16 v16bf;
typedef __attribute__((ext_vector_type(8)))  float  v8f;
typedef __attribute__((ext_vector_type(4)))  float  f32x4;
typedef __attribute__((ext_vector_type(4)))  unsigned int u32x4;

union FragU { u32x4 q[2]; v16bf v; };

__device__ __forceinline__ unsigned short f2bf(float f) {
  unsigned int u = __float_as_uint(f);
  u += 0x7FFFu + ((u >> 16) & 1u);      // round-to-nearest-even truncation
  return (unsigned short)(u >> 16);
}
// order-preserving float <-> uint map for atomic max
__device__ __forceinline__ unsigned int fmap(float f) {
  unsigned int u = __float_as_uint(f);
  return (u & 0x80000000u) ? ~u : (u | 0x80000000u);
}
__device__ __forceinline__ float funmap(unsigned int u) {
  u = (u & 0x80000000u) ? (u & 0x7FFFFFFFu) : ~u;
  return __uint_as_float(u);
}

// ======================================================================
// WMMA GEMM:  C[M,256] = A[M,K](f32, row major)  x  BT[256,K](bf16)^T
// block = 256 threads (8 waves). Block tile 32(M) x 256(N).
// wave w -> N offset w*32; each wave: 2x2 tiles of 16x16, K step 32.
// M, K multiples of 32 (load clamp + store guard kept for safety).
// ======================================================================
#define LDS_STRIDE 40   // ushorts per 32-wide row (+8 pad -> conflict-free b128)

__global__ __launch_bounds__(256)
void k_gemm_bf16_n256(const float* __restrict__ A,
                      const unsigned short* __restrict__ BT,
                      float* __restrict__ C, int M, int K)
{
  __shared__ unsigned short sA[32  * LDS_STRIDE];
  __shared__ unsigned short sB[256 * LDS_STRIDE];

  const int tid  = threadIdx.x;
  const int wave = tid >> 5;
  const int lane = tid & 31;
  const int half = lane >> 4;
  const int r16  = lane & 15;
  const int m0   = blockIdx.x * 32;
  const int kb   = half * 8;            // per-lane K base inside 32-chunk

  v8f acc[2][2] = {};

  for (int k0 = 0; k0 < K; k0 += 32) {
    // ---- stage A tile 32x32 f32 -> bf16 LDS (1 float4 per thread) ----
    {
      int row = tid >> 3, c4 = (tid & 7) << 2;
      int gr  = m0 + row; if (gr > M - 1) gr = M - 1;
      f32x4 av = *reinterpret_cast<const f32x4*>(A + (size_t)gr * K + k0 + c4);
      unsigned short* d = &sA[row * LDS_STRIDE + c4];
      d[0] = f2bf(av.x); d[1] = f2bf(av.y); d[2] = f2bf(av.z); d[3] = f2bf(av.w);
    }
    // ---- stage B tile 256x16 bf16 (first 32B of one row per thread) ----
    {
      const u32x4* s = reinterpret_cast<const u32x4*>(BT + (size_t)tid * K + k0);
      u32x4* d = reinterpret_cast<u32x4*>(&sB[tid * LDS_STRIDE]);
      d[0] = s[0]; d[1] = s[1];
    }
    __syncthreads();

    FragU fa[2], fb[2];
#pragma unroll
    for (int t = 0; t < 2; ++t) {
      const unsigned short* p = &sA[(t * 16 + r16) * LDS_STRIDE + kb];
      fa[t].q[0] = *reinterpret_cast<const u32x4*>(p);
      fa[t].q[1] = *reinterpret_cast<const u32x4*>(p + 16);
    }
#pragma unroll
    for (int u = 0; u < 2; ++u) {
      const unsigned short* p = &sB[(wave * 32 + u * 16 + r16) * LDS_STRIDE + kb];
      fb[u].q[0] = *reinterpret_cast<const u32x4*>(p);
      fb[u].q[1] = *reinterpret_cast<const u32x4*>(p + 16);
    }
#pragma unroll
    for (int t = 0; t < 2; ++t)
#pragma unroll
      for (int u = 0; u < 2; ++u)
        acc[t][u] = __builtin_amdgcn_wmma_f32_16x16x32_bf16(
            false, fa[t].v, false, fb[u].v, (short)0, acc[t][u], false, false);
    __syncthreads();
  }

#pragma unroll
  for (int t = 0; t < 2; ++t)
#pragma unroll
    for (int u = 0; u < 2; ++u) {
      int col = wave * 32 + u * 16 + r16;
#pragma unroll
      for (int e = 0; e < 8; ++e) {
        int row = m0 + t * 16 + e + half * 8;
        if (row < M) C[(size_t)row * 256 + col] = acc[t][u][e];
      }
    }
}

// ---------------- W -> WT (bf16, transposed) ----------------
__global__ void k_conv_WT(const float* __restrict__ W, unsigned short* __restrict__ WT) {
  int t = blockIdx.x * blockDim.x + threadIdx.x;      // 256*64
  if (t >= HCC * HIDDENC) return;
  int n = t >> 6, k = t & 63;
  WT[n * HIDDENC + k] = f2bf(W[(size_t)k * HCC + n]);
}

// ---------------- per-node attention logits ----------------
__global__ void k_att(const float* __restrict__ h, const float* __restrict__ att_s,
                      const float* __restrict__ att_d,
                      float* __restrict__ asrc, float* __restrict__ adst) {
  int t = blockIdx.x * blockDim.x + threadIdx.x;
  if (t >= N_NODESC * HEADSC) return;
  int n = t >> 2, hd = t & 3;
  const float* hp = h + (size_t)n * HCC + hd * HIDDENC;
  const float* a1 = att_s + hd * HIDDENC;
  const float* a2 = att_d + hd * HIDDENC;
  float s1 = 0.f, s2 = 0.f;
#pragma unroll
  for (int c = 0; c < HIDDENC; c += 4) {
    f32x4 hv = *reinterpret_cast<const f32x4*>(hp + c);
    f32x4 v1 = *reinterpret_cast<const f32x4*>(a1 + c);
    f32x4 v2 = *reinterpret_cast<const f32x4*>(a2 + c);
    s1 += hv.x * v1.x + hv.y * v1.y + hv.z * v1.z + hv.w * v1.w;
    s2 += hv.x * v2.x + hv.y * v2.y + hv.z * v2.z + hv.w * v2.w;
  }
  asrc[t] = s1; adst[t] = s2;
}

// ---------------- zero scratch accumulators ----------------
__global__ void k_zero(float* __restrict__ p, int n) {
  int t = blockIdx.x * blockDim.x + threadIdx.x;
  if (t < n) p[t] = 0.0f;
}

__device__ __forceinline__ void edge_sd(const int* ei, int i, int& s, int& d) {
  if (i < N_EDGESC) { s = ei[i]; d = ei[N_EDGESC + i]; }
  else              { s = d = i - N_EDGESC; }
}

// ---------------- pass 1: leaky-relu score, segment max; emit edge_index_sl ----
__global__ void k_edge_max(const int* __restrict__ ei, const float* __restrict__ asrc,
                           const float* __restrict__ adst, unsigned int* __restrict__ emax,
                           int* __restrict__ out_ei) {
  int t = blockIdx.x * blockDim.x + threadIdx.x;
  if (t >= E_TOTC * HEADSC) return;
  int i = t >> 2, hd = t & 3, s, d;
  edge_sd(ei, i, s, d);
  float e = asrc[s * 4 + hd] + adst[d * 4 + hd];
  e = e > 0.f ? e : NEG_SLOPEC * e;
  atomicMax(&emax[d * 4 + hd], fmap(e));
  if (hd == 0) { out_ei[i] = s; out_ei[E_TOTC + i] = d; }
}

// ---------------- pass 2: exp(e - max), segment sum ----------------
__global__ void k_edge_exp(const int* __restrict__ ei, const float* __restrict__ asrc,
                           const float* __restrict__ adst, const unsigned int* __restrict__ emax,
                           float* __restrict__ eexp, float* __restrict__ esum) {
  int t = blockIdx.x * blockDim.x + threadIdx.x;
  if (t >= E_TOTC * HEADSC) return;
  int i = t >> 2, hd = t & 3, s, d;
  edge_sd(ei, i, s, d);
  float e = asrc[s * 4 + hd] + adst[d * 4 + hd];
  e = e > 0.f ? e : NEG_SLOPEC * e;
  float ex = expf(e - funmap(emax[d * 4 + hd]));
  eexp[t] = ex;
  atomicAdd(&esum[d * 4 + hd], ex);
}

// ---------------- pass 3: alpha ----------------
__global__ void k_edge_alpha(const int* __restrict__ ei, const float* __restrict__ eexp,
                             const float* __restrict__ esum, float* __restrict__ out_alpha) {
  int t = blockIdx.x * blockDim.x + threadIdx.x;
  if (t >= E_TOTC * HEADSC) return;
  int i = t >> 2, hd = t & 3, s, d;
  edge_sd(ei, i, s, d);
  out_alpha[t] = eexp[t] / (esum[d * 4 + hd] + 1e-16f);
}

// ---------------- weighted message aggregation (f32 atomics) ----------------
__global__ void k_aggregate(const int* __restrict__ ei, const float* __restrict__ h,
                            const float* __restrict__ alpha, float* __restrict__ gat) {
  int t = blockIdx.x * blockDim.x + threadIdx.x;    // E_TOT * 64 threads
  if (t >= E_TOTC * 64) return;
  int i = t >> 6, j0 = (t & 63) << 2, hd = j0 >> 6, s, d;
  edge_sd(ei, i, s, d);
  float al = alpha[i * 4 + hd];
  f32x4 hv = *reinterpret_cast<const f32x4*>(h + (size_t)s * HCC + j0);
  float* o = gat + (size_t)d * HCC + j0;
  atomicAdd(o + 0, hv.x * al);
  atomicAdd(o + 1, hv.y * al);
  atomicAdd(o + 2, hv.z * al);
  atomicAdd(o + 3, hv.w * al);
}

// ---------------- bias + ELU + transpose to bf16 [256][20000] ----------------
__global__ void k_bias_elu_T(const float* __restrict__ gat, const float* __restrict__ bias,
                             unsigned short* __restrict__ hT) {
  int n = blockIdx.x, j = threadIdx.x;
  float v = gat[(size_t)n * HCC + j] + bias[j];
  float e = v > 0.f ? v : expm1f(v);
  hT[(size_t)j * N_NODESC + n] = f2bf(e);
}

// ---------------- fc1 (relu) ----------------
__global__ void k_fc1(const float* __restrict__ sr, const float* __restrict__ w,
                      const float* __restrict__ b, float* __restrict__ o1) {
  int t = blockIdx.x * blockDim.x + threadIdx.x;    // 4096*256
  if (t >= BATCHC * 256) return;
  int bi = t >> 8, j = t & 255;
  const float* xp = sr + (size_t)bi * 256;
  const float* wp = w + (size_t)j * 256;
  float s = b[j];
#pragma unroll 4
  for (int c = 0; c < 256; c += 4) {
    f32x4 xv = *reinterpret_cast<const f32x4*>(xp + c);
    f32x4 wv = *reinterpret_cast<const f32x4*>(wp + c);
    s += xv.x * wv.x + xv.y * wv.y + xv.z * wv.z + xv.w * wv.w;
  }
  o1[t] = fmaxf(s, 0.0f);
}

// ---------------- fc2 ----------------
__global__ void k_fc2(const float* __restrict__ o1, const float* __restrict__ w,
                      const float* __restrict__ b, float* __restrict__ o) {
  int t = blockIdx.x * blockDim.x + threadIdx.x;    // 4096*10
  if (t >= BATCHC * NCLSC) return;
  int bi = t / NCLSC, c = t % NCLSC;
  const float* xp = o1 + (size_t)bi * 256;
  const float* wp = w + (size_t)c * 256;
  float s = b[c];
#pragma unroll 4
  for (int k = 0; k < 256; k += 4) {
    f32x4 xv = *reinterpret_cast<const f32x4*>(xp + k);
    f32x4 wv = *reinterpret_cast<const f32x4*>(wp + k);
    s += xv.x * wv.x + xv.y * wv.y + xv.z * wv.z + xv.w * wv.w;
  }
  o[t] = s;
}

// ======================================================================
extern "C" void kernel_launch(void* const* d_in, const int* in_sizes, int n_in,
                              void* d_out, int out_size, void* d_ws, size_t ws_size,
                              hipStream_t stream)
{
  const float* x     = (const float*)d_in[0];
  const int*   ei    = (const int*)  d_in[1];
  const float* feat  = (const float*)d_in[2];
  const float* W     = (const float*)d_in[3];
  const float* att_s = (const float*)d_in[4];
  const float* att_d = (const float*)d_in[5];
  const float* bias  = (const float*)d_in[6];
  const float* fc1w  = (const float*)d_in[7];
  const float* fc1b  = (const float*)d_in[8];
  const float* fc2w  = (const float*)d_in[9];
  const float* fc2b  = (const float*)d_in[10];
  (void)in_sizes; (void)n_in; (void)out_size; (void)ws_size;

  // ---- workspace layout (bytes, all 16B aligned) ----
  char* ws = (char*)d_ws;
  float*          h    = (float*)         (ws + 0);          // 20,480,000
  float*          asrc = (float*)         (ws + 20480000);   //    320,000
  float*          adst = (float*)         (ws + 20800000);   //    320,000
  float*          eexp = (float*)         (ws + 21120000);   // 10,560,000
  unsigned short* hT   = (unsigned short*)(ws + 31680000);   // 10,240,000
  unsigned short* WT   = (unsigned short*)(ws + 41920000);   //     32,768
  float*          sr   = (float*)         (ws + 41952768);   //  4,194,304
  float*          o1   = (float*)         (ws + 46147072);   //  4,194,304
  unsigned int*   emax = (unsigned int*)  (ws + 50341376);   //    320,000
  float*          esum = (float*)         (ws + 50661376);   //    320,000
  float*          gat  = (float*)         (ws + 50981376);   // 20,480,000

  // ---- output layout: [o (40960 f32)] [edge_index_sl (2*660000 i32)] [alpha (660000*4 f32)]
  float* out_o     = (float*)d_out;
  int*   out_ei    = (int*)d_out + BATCHC * NCLSC;
  float* out_alpha = (float*)d_out + BATCHC * NCLSC + 2 * E_TOTC;

  const int EH  = E_TOTC * HEADSC;                 // 2,640,000
  const int EHB = (EH + 255) / 256;

  // 1) WT = bf16(W^T)
  k_conv_WT<<<(HCC * HIDDENC) / 256, 256, 0, stream>>>(W, WT);
  // 2) h = feat_embed @ W  (WMMA bf16, M=20000, K=64)
  k_gemm_bf16_n256<<<N_NODESC / 32, 256, 0, stream>>>(feat, WT, h, N_NODESC, HIDDENC);
  // 3) attention logits
  k_att<<<(N_NODESC * HEADSC + 255) / 256, 256, 0, stream>>>(h, att_s, att_d, asrc, adst);
  // 4) zero emax/esum/gat (contiguous region)
  k_zero<<<(320000 + 320000 + 20480000) / 4 / 256, 256, 0, stream>>>((float*)emax, 5280000);
  // 5) segment max (+ emit edge_index_sl)
  k_edge_max<<<EHB, 256, 0, stream>>>(ei, asrc, adst, emax, out_ei);
  // 6) exp + segment sum
  k_edge_exp<<<EHB, 256, 0, stream>>>(ei, asrc, adst, emax, eexp, esum);
  // 7) alpha
  k_edge_alpha<<<EHB, 256, 0, stream>>>(ei, eexp, esum, out_alpha);
  // 8) aggregation: gat[d] += alpha * h[s]
  k_aggregate<<<E_TOTC * 64 / 256, 256, 0, stream>>>(ei, h, out_alpha, gat);
  // 9) bias + ELU, transpose to bf16 [256][20000]
  k_bias_elu_T<<<N_NODESC, 256, 0, stream>>>(gat, bias, hT);
  // 10) sample_repr = x @ h_out  (WMMA bf16, M=4096, K=20000) — dominant GEMM
  k_gemm_bf16_n256<<<BATCHC / 32, 256, 0, stream>>>(x, hT, sr, BATCHC, N_NODESC);
  // 11) fc1 + relu
  k_fc1<<<BATCHC * 256 / 256, 256, 0, stream>>>(sr, fc1w, fc1b, o1);
  // 12) fc2
  k_fc2<<<(BATCHC * NCLSC + 255) / 256, 256, 0, stream>>>(o1, fc2w, fc2b, out_o);
}
